// AGRU_55662776156443
// MI455X (gfx1250) — matrix operand order repaired and checked by reading, not compiled, not run on hardware
//
#include <hip/hip_runtime.h>
#include <hip/hip_bf16.h>

// AGRU on MI455X (gfx1250, wave32):
//  - one-time bf16 conversion of x / w_ih / w_hh (memory-bound pre-pass)
//  - gx = x @ w_ih^T + b_ih : parallel GEMM, all-bf16 operands, f32 accum WMMA
//  - 200-step recurrence: gh = h @ w_hh^T via WMMA + fused gate math;
//    h kept in f32 (exact gating) AND bf16 (WMMA A operand) to keep the
//    sequential inner loop free of cvt instructions.

typedef __attribute__((ext_vector_type(16))) __bf16 v16bf;
typedef __attribute__((ext_vector_type(8)))  __bf16 v8bf;
typedef __attribute__((ext_vector_type(4)))  __bf16 v4bf;
typedef __attribute__((ext_vector_type(8)))  float  v8f;

#define B_  512
#define L_  200
#define I_  512
#define H_  512
#define G_  1536   // 3*H

// ---- per-lane WMMA operand loaders (wave32 layouts, CDNA5 ISA) ----

// 16-bit A matrix 16x32 (MxK): lane holds row M = lane%16.
// lanes 0-15: v0-3 = K0..7, v4-7 = K16..23 ; lanes 16-31: K8..15 and K24..31.
__device__ __forceinline__ v16bf load_a_bf16(const __bf16* __restrict__ row,
                                             int kbase, int lane) {
  const int khalf = ((lane >> 4) & 1) * 8;
  v8bf a0 = *(const v8bf*)(row + kbase + khalf);        // one b128
  v8bf a1 = *(const v8bf*)(row + kbase + 16 + khalf);   // one b128
  v16bf a;
#pragma unroll
  for (int i = 0; i < 8; ++i) { a[i] = a0[i]; a[8 + i] = a1[i]; }
  return a;
}

// 16-bit B matrix 32x16 (KxN): lane holds column N = lane%16.
// lanes 0-15: K0..7 and K16..23; lanes 16-31: K8..15 and K24..31 (same as A).
// B[k][n] = W[n][k], so `row` points at bf16 weight row n (contiguous K).
__device__ __forceinline__ v16bf load_b_bf16(const __bf16* __restrict__ row,
                                             int kbase, int lane) {
  const int k0 = kbase + ((lane & 16) ? 8 : 0);
  v8bf b0 = *(const v8bf*)(row + k0);
  v8bf b1 = *(const v8bf*)(row + k0 + 16);
  v16bf b;
#pragma unroll
  for (int i = 0; i < 8; ++i) { b[i] = b0[i]; b[8 + i] = b1[i]; }
  return b;
}

// ---- vectorized one-time f32 -> bf16 conversion (4 elems / thread) ----
__global__ void cvt_f32_bf16_v4(const float* __restrict__ src,
                                __bf16* __restrict__ dst, int n4) {
  int i = blockIdx.x * blockDim.x + threadIdx.x;
  if (i < n4) {
    float4 f = ((const float4*)src)[i];
    v4bf o;
    o[0] = (__bf16)f.x; o[1] = (__bf16)f.y; o[2] = (__bf16)f.z; o[3] = (__bf16)f.w;
    ((v4bf*)dst)[i] = o;
  }
}

// ---- gx_all[l][b][g] = sum_i x[b][l][i] * w_ih[g][i] + b_ih[g] ----
// M = L*B = 102400 rows (r = l*B + b), N = G = 1536, K = I = 512.
// Block: 64(M) x 32(N), 8 waves, one 16x16 tile per wave.
__global__ void gx_gemm(const __bf16* __restrict__ xb, const __bf16* __restrict__ wih,
                        const float* __restrict__ b_ih, float* __restrict__ gx) {
  const int lane   = threadIdx.x & 31;
  const int wave   = threadIdx.x >> 5;
  const int m_base = blockIdx.x * 64 + (wave >> 1) * 16;
  const int n_base = blockIdx.y * 32 + (wave & 1) * 16;

  // A row for this lane: global row r = l*B + b  ->  x[b][l][:]
  const int r_row = m_base + (lane & 15);
  const int l_idx = r_row >> 9;          // / 512
  const int b_idx = r_row & 511;         // % 512
  const __bf16* arow = xb + ((size_t)b_idx * L_ + l_idx) * I_;

  const int ncol = n_base + (lane & 15); // weight row == output column
  const __bf16* brow = wih + (size_t)ncol * I_;

  v8f acc = {};
#pragma unroll 4
  for (int k = 0; k < I_; k += 32) {
    v16bf a = load_a_bf16(arow, k, lane);
    v16bf b = load_b_bf16(brow, k, lane);
    acc = __builtin_amdgcn_wmma_f32_16x16x32_bf16(false, a, false, b,
                                                  (short)0, acc, false, false);
  }

  // C layout: lane holds column ncol; VGPR v -> row m_base + v (+8 for lanes 16-31)
  const float bias = b_ih[ncol];
  const int mv = m_base + ((lane >> 4) & 1) * 8;
#pragma unroll
  for (int v = 0; v < 8; ++v)
    gx[(size_t)(mv + v) * G_ + ncol] = acc[v] + bias;
}

// ---- one recurrence step: gh = h_prev @ w_hh^T, fused gating ----
// M = B = 512 rows, N covers H=512 columns x 3 gates; same (m,n) tile per gate
// so r,z,n land in the same lanes for the epilogue. Block: 64 x 32, 8 waves.
__global__ void agru_step(int l,
                          const float*  __restrict__ h_prev,   // f32 state (exact)
                          const __bf16* __restrict__ hb_prev,  // bf16 mirror (WMMA A)
                          float*  __restrict__ h_next,
                          __bf16* __restrict__ hb_next,
                          const __bf16* __restrict__ whh, const float* __restrict__ b_hh,
                          const float* __restrict__ gx, const float* __restrict__ att,
                          float* __restrict__ out) {
  const int lane   = threadIdx.x & 31;
  const int wave   = threadIdx.x >> 5;
  const int m_base = blockIdx.x * 64 + (wave >> 1) * 16;
  const int n_base = blockIdx.y * 32 + (wave & 1) * 16;

  const __bf16* arow = hb_prev + (size_t)(m_base + (lane & 15)) * H_;
  const int ncol = n_base + (lane & 15);
  const __bf16* br = whh + (size_t)(0 * H_ + ncol) * H_;   // gate r row
  const __bf16* bz = whh + (size_t)(1 * H_ + ncol) * H_;   // gate z row
  const __bf16* bn = whh + (size_t)(2 * H_ + ncol) * H_;   // gate n row

  v8f ar = {}, az = {}, an = {};
#pragma unroll 4
  for (int k = 0; k < H_; k += 32) {
    v16bf a  = load_a_bf16(arow, k, lane);                 // shared A tile
    v16bf wr = load_b_bf16(br, k, lane);
    v16bf wz = load_b_bf16(bz, k, lane);
    v16bf wn = load_b_bf16(bn, k, lane);
    ar = __builtin_amdgcn_wmma_f32_16x16x32_bf16(false, a, false, wr, (short)0, ar, false, false);
    az = __builtin_amdgcn_wmma_f32_16x16x32_bf16(false, a, false, wz, (short)0, az, false, false);
    an = __builtin_amdgcn_wmma_f32_16x16x32_bf16(false, a, false, wn, (short)0, an, false, false);
  }

  const float bhr = b_hh[ncol];
  const float bhz = b_hh[ncol + H_];
  const float bhn = b_hh[ncol + 2 * H_];
  const int mv = m_base + ((lane >> 4) & 1) * 8;

#pragma unroll
  for (int v = 0; v < 8; ++v) {
    const int m = mv + v;                                  // batch index b
    const size_t gxo = ((size_t)l * B_ + m) * G_ + ncol;
    const float ghr = ar[v] + bhr;
    const float ghz = az[v] + bhz;
    const float ghn = an[v] + bhn;
    const float r  = 1.0f / (1.0f + __expf(-(gx[gxo]          + ghr)));
    const float z  = 1.0f / (1.0f + __expf(-(gx[gxo + H_]     + ghz)));
    const float nn = tanhf(gx[gxo + 2 * H_] + r * ghn);
    const float h  = h_prev[(size_t)m * H_ + ncol];
    const float ho = (1.0f - z) * nn + z * h;
    const float w  = att[(size_t)l * B_ + m];
    const float hv = (1.0f - w) * h + w * ho;
    h_next [(size_t)m * H_ + ncol] = hv;
    hb_next[(size_t)m * H_ + ncol] = (__bf16)hv;           // next step's A operand
    out[((size_t)m * L_ + l) * H_ + ncol] = hv;            // out[b][l][n]
  }
}

extern "C" void kernel_launch(void* const* d_in, const int* in_sizes, int n_in,
                              void* d_out, int out_size, void* d_ws, size_t ws_size,
                              hipStream_t stream) {
  const float* x    = (const float*)d_in[0];
  const float* att  = (const float*)d_in[1];
  const float* h0   = (const float*)d_in[2];
  const float* w_ih = (const float*)d_in[3];
  const float* w_hh = (const float*)d_in[4];
  const float* b_ih = (const float*)d_in[5];
  const float* b_hh = (const float*)d_in[6];
  float* out = (float*)d_out;

  // ---- workspace layout (all offsets large-power-of-two aligned) ----
  char* ws = (char*)d_ws;
  const size_t gx_bytes = (size_t)L_ * B_ * G_ * sizeof(float);    // ~630 MB
  const size_t w_bytes  = (size_t)G_ * I_ * sizeof(__bf16);        // 1.5 MB each
  const size_t xb_bytes = (size_t)B_ * L_ * I_ * sizeof(__bf16);   // 100 MB
  const size_t h_bytes  = (size_t)B_ * H_ * sizeof(float);         // 1 MB
  const size_t hb_bytes = (size_t)B_ * H_ * sizeof(__bf16);        // 0.5 MB

  size_t off = 0;
  float*  gx    = (float*)(ws + off);  off += gx_bytes;
  __bf16* wihb  = (__bf16*)(ws + off); off += w_bytes;
  __bf16* whhb  = (__bf16*)(ws + off); off += w_bytes;
  __bf16* xb    = (__bf16*)(ws + off); off += xb_bytes;
  float*  hbuf0 = (float*)(ws + off);  off += h_bytes;
  float*  hbuf1 = (float*)(ws + off);  off += h_bytes;
  __bf16* hbf0  = (__bf16*)(ws + off); off += hb_bytes;
  __bf16* hbf1  = (__bf16*)(ws + off); off += hb_bytes;

  // ---- h <- h0 (f32 and bf16 mirrors) ----
  hipMemcpyAsync(hbuf0, h0, h_bytes, hipMemcpyDeviceToDevice, stream);
  cvt_f32_bf16_v4<<<(B_ * H_ / 4 + 255) / 256, 256, 0, stream>>>(h0, hbf0, B_ * H_ / 4);

  // ---- one-time bf16 conversions (memory-bound pre-pass) ----
  const int nw4 = G_ * I_ / 4;
  cvt_f32_bf16_v4<<<(nw4 + 255) / 256, 256, 0, stream>>>(w_ih, wihb, nw4);
  cvt_f32_bf16_v4<<<(nw4 + 255) / 256, 256, 0, stream>>>(w_hh, whhb, nw4);
  const int nx4 = B_ * L_ * I_ / 4;
  cvt_f32_bf16_v4<<<(nx4 + 255) / 256, 256, 0, stream>>>(x, xb, nx4);

  // ---- parallel input-projection GEMM: (L*B x G, K=I) ----
  gx_gemm<<<dim3((L_ * B_) / 64, G_ / 32), 256, 0, stream>>>(xb, wihb, b_ih, gx);

  // ---- sequential recurrence: one graph node per step ----
  float*  hp  = hbuf0; float*  hn  = hbuf1;
  __bf16* hpb = hbf0;  __bf16* hnb = hbf1;
  for (int l = 0; l < L_; ++l) {
    agru_step<<<dim3(B_ / 64, H_ / 32), 256, 0, stream>>>(l, hp, hpb, hn, hnb,
                                                          whhb, b_hh, gx, att, out);
    float*  tf = hp;  hp  = hn;  hn  = tf;
    __bf16* tb = hpb; hpb = hnb; hnb = tb;
  }

  // ---- hx = final hidden state (second output, after out) ----
  hipMemcpyAsync(out + (size_t)B_ * L_ * H_, hp, h_bytes,
                 hipMemcpyDeviceToDevice, stream);
}
